// DeformableAttnBlock_FUSION_45973329937040
// MI455X (gfx1250) — matrix operand order repaired and checked by reading, not compiled, not run on hardware
//
#include <hip/hip_runtime.h>
#include <stdint.h>

// ---------------------------------------------------------------------------
// CDNA5 / gfx1250 implementation of DeformableAttnBlock_FUSION
// - All convolutions: implicit-GEMM with v_wmma_f32_16x16x32_bf16
//   (bf16 activations/weights, f32 accumulate), fully unrolled taps,
//   2 N-subtiles per wave so each A fragment feeds two WMMAs (LDS BW relief).
// - Weights pre-packed to bf16 in LDS-tile order; per-block weight tiles are
//   fetched with the Tensor Data Mover (TENSOR_LOAD_TO_LDS + s_wait_tensorcnt)
//   into ping-pong LDS buffers, overlapped with WMMA compute.
// - Gather stages (warp, deform sampling, softmax): VALU kernels, L2-resident.
// ---------------------------------------------------------------------------

typedef __attribute__((ext_vector_type(16))) __bf16 v16bf;
typedef __attribute__((ext_vector_type(8)))  float  v8f;
typedef unsigned int __attribute__((ext_vector_type(4))) v4u;
typedef int __attribute__((ext_vector_type(4)))          v4i;
typedef int __attribute__((ext_vector_type(8)))          v8i;

#if defined(__HIP_DEVICE_COMPILE__) && __has_builtin(__builtin_amdgcn_tensor_load_to_lds)
#define USE_TDM 1
#else
#define USE_TDM 0
#endif

// ---------------------------------------------------------------------------
// Pre-pack conv weights (OIHW f32) -> bf16 tile layout:
//   wpk[co][chunk][tap*32 + cl],  chunk = ci/32, tap = r*KW+s, cl = ci%32
// Zero-pads Cin up to a multiple of 32 so the GEMM K-tail needs no guards.
// ---------------------------------------------------------------------------
__global__ void k_pack_weights(const float* __restrict__ w, __bf16* __restrict__ wpk,
                               int Cin, int Cout, int KH, int KW, long total)
{
    long i = (long)blockIdx.x * blockDim.x + threadIdx.x;
    if (i >= total) return;
    const int taps = KH * KW;
    const int nch  = (Cin + 31) >> 5;
    int cl  = (int)(i & 31);
    long t2 = i >> 5;
    int tap = (int)(t2 % taps);
    long t3 = t2 / taps;
    int ch  = (int)(t3 % nch);
    int co  = (int)(t3 / nch);
    int ci  = ch * 32 + cl;
    float v = 0.0f;
    if (ci < Cin) {
        int r = tap / KW, s = tap - r * KW;
        v = w[(((long)co * Cin + ci) * KH + r) * KW + s];
    }
    wpk[i] = (__bf16)v;
}

// ---------------------------------------------------------------------------
// Implicit-GEMM conv (KHxKW compile-time, pad/dilation runtime), WMMA bf16.
// Block = 256 threads = 8 waves. Block tile: 16 Cout x 256 contiguous pixels
// (one image row; W % 256 == 0). Each wave: 16 Cout x 32 px (two 16x16 WMMA
// accumulators sharing one A fragment per tap).
// K chunked 32 input channels at a time through ping-pong LDS buffers:
//   chunk c+1 staged (TDM DMA weights + manual im2col) while chunk c computes.
// ---------------------------------------------------------------------------
template<int KH, int KW>
__global__ __launch_bounds__(256) void conv_wmma_kernel(
    const __bf16* __restrict__ in, int Cin, long in_bstride,
    const __bf16* __restrict__ wpk, const float* __restrict__ bias, int Cout, int nch,
    int pad, int dil,
    float* __restrict__ outf, __bf16* __restrict__ outbf, long out_bstride,
    const float* __restrict__ res, long res_bstride,
    int H, int W, int leaky)
{
    constexpr int TAPS = KH * KW;
    constexpr int PWS  = 260;                  // sX pixel stride (>= 256 + 2*dil)
    const int HW = H * W;
    __shared__ __bf16 sX[2][KH * PWS * 32];    // im2col patch (ping-pong)
    __shared__ __bf16 sW[2][16 * TAPS * 32];   // weight tile  (ping-pong)

    const int tid  = threadIdx.x;
    const int lane = tid & 31;
    const int wv   = tid >> 5;
    const int bz   = blockIdx.z;
    const int co0  = blockIdx.y * 16;
    const int Q0   = blockIdx.x * 256;
    const int y    = Q0 / W;
    const int x0   = Q0 % W;
    const int xs   = wv * 32;                  // this wave's 32-pixel sub-tile
    const int PW   = 256 + (KW - 1) * dil;
    const int rowlen = nch * TAPS * 32;        // packed-weight row length per Cout

    const __bf16* inb = in + (long)bz * in_bstride;

    v8f acc0 = {}, acc1 = {};

    // ISA 7.12.2 fragment geometry (wave32):
    const int mA  = lane & 15;                 // A: lane -> M (Cout row)
    const int nB  = lane & 15;                 // B/D: lane -> N (pixel)
    const int kbA = (lane >> 4) * 8;           // A K-base (lanes 16-31: K=8..15/24..31)
    const int kbB = (lane >> 4) * 8;           // B K-base (lanes 16-31: K=8..15/24..31)

    // ---- stage weight tile (16 rows x TAPS*32 bf16) into sW[buf]
    auto stage_w = [&](int ci0, __bf16* sWbuf) {
#if USE_TDM
        if (wv == 0) {
            unsigned long long ga = (unsigned long long)(uintptr_t)
                (wpk + ((long)co0 * nch + (ci0 >> 5)) * (long)(TAPS * 32));
            v4u g0;
            g0.x = 1u;                                        // count=1 (valid user D#)
            g0.y = (unsigned int)(uintptr_t)sWbuf;            // lds_addr = addr[31:0]
            g0.z = (unsigned int)ga;                          // global_addr[31:0]
            g0.w = (unsigned int)((ga >> 32) & 0x01FFFFFFull) // global_addr[56:32]
                 | (2u << 30);                                // type = 2 ("image")
            // D# group1: direct bitfield packing (offsets per ISA 8.4)
            union { unsigned int u[8]; v8i v; } g1;
            const unsigned int rl = (unsigned int)rowlen;
            g1.u[0] = 1u << 16;                               // data_size = 2 bytes
            g1.u[1] = (rl & 0xFFFFu) << 16;                   // tensor_dim0[15:0]  @48
            g1.u[2] = ((rl >> 16) & 0xFFFFu)                  // tensor_dim0[31:16]
                    | (((unsigned int)Cout & 0xFFFFu) << 16); // tensor_dim1[15:0]  @80
            g1.u[3] = (((unsigned int)Cout >> 16) & 0xFFFFu)  // tensor_dim1[31:16]
                    | ((unsigned int)(TAPS * 32) << 16);      // tile_dim0          @112
            g1.u[4] = 16u;                                    // tile_dim1 = 16     @128
            g1.u[5] = rl;                                     // dim0_stride[31:0]  @160
            g1.u[6] = 0u;                                     // dim0_stride[47:32], dim1_stride lo
            g1.u[7] = 0u;
            v4i z4 = 0;
#if __clang_major__ >= 23
            v8i z8 = 0;
            __builtin_amdgcn_tensor_load_to_lds(g0, g1.v, z4, z4, z8, 0);
#else
            __builtin_amdgcn_tensor_load_to_lds(g0, g1.v, z4, z4, 0);
#endif
        }
#else
        const __bf16* wsrc = wpk + ((long)co0 * nch + (ci0 >> 5)) * (long)(TAPS * 32);
        for (int idx = tid; idx < 16 * TAPS * 32; idx += 256) {
            int m   = idx / (TAPS * 32);
            int rem = idx - m * (TAPS * 32);
            sWbuf[idx] = wsrc[(long)m * rowlen + rem];
        }
#endif
    };

    // ---- stage im2col patch: sXbuf[(r*PWS + xl)*32 + cl] (channel-innermost)
    auto stage_x = [&](int ci0, __bf16* sXbuf) {
        const int xElems = KH * PW * 32;
        for (int idx = tid; idx < xElems; idx += 256) {
            int cl  = idx & 31;
            int pos = idx >> 5;
            int r   = pos / PW;
            int xl  = pos - r * PW;
            int ci  = ci0 + cl;
            int ys  = y + r * dil - pad;
            int xsrc = x0 + xl - pad;
            __bf16 v = (__bf16)0.0f;
            if (ci < Cin && (unsigned)ys < (unsigned)H && (unsigned)xsrc < (unsigned)W)
                v = inb[(long)ci * HW + (long)ys * W + xsrc];
            sXbuf[(r * PWS + xl) * 32 + cl] = v;
        }
    };

    // ---- prologue: stage chunk 0 into buffer 0
    stage_w(0, sW[0]);
    stage_x(0, sX[0]);
#if USE_TDM
    if (wv == 0) __builtin_amdgcn_s_wait_tensorcnt(0);
#endif
    __syncthreads();

    int cur = 0;
    for (int c = 0; c < nch; ++c) {
        // stage next chunk into the other buffer (TDM DMA overlaps compute)
        if (c + 1 < nch) {
            __builtin_prefetch(inb + (long)(c + 1) * 32 * HW + (long)y * W + x0, 0, 1);
            stage_w((c + 1) * 32, sW[cur ^ 1]);
            stage_x((c + 1) * 32, sX[cur ^ 1]);
        }

        // ---- GEMM: per tap, one A fragment feeds two 16x16x32 WMMAs
        const __bf16* sWc = sW[cur];
        const __bf16* sXc = sX[cur];
        #pragma unroll
        for (int r = 0; r < KH; ++r) {
            #pragma unroll
            for (int s = 0; s < KW; ++s) {
                union { uint32_t u[8]; v16bf v; } Af, B0, B1;
                const __bf16* wrow = &sWc[mA * (TAPS * 32) + (r * KW + s) * 32];
                #pragma unroll
                for (int vv = 0; vv < 8; ++vv) {
                    int ka = kbA + ((vv < 4) ? (2 * vv) : (2 * vv + 8));
                    Af.u[vv] = *(const uint32_t*)(wrow + ka);
                }
                const __bf16* xrow = &sXc[(r * PWS + xs + nB + s * dil) * 32 + kbB];
                #pragma unroll
                for (int vv = 0; vv < 8; ++vv) {
                    int kb = (vv < 4) ? (2 * vv) : (2 * vv + 8);
                    B0.u[vv] = *(const uint32_t*)(xrow + kb);
                    B1.u[vv] = *(const uint32_t*)(xrow + 16 * 32 + kb);
                }
                acc0 = __builtin_amdgcn_wmma_f32_16x16x32_bf16(
                    false, Af.v, false, B0.v, (short)0, acc0, false, false);
                acc1 = __builtin_amdgcn_wmma_f32_16x16x32_bf16(
                    false, Af.v, false, B1.v, (short)0, acc1, false, false);
            }
        }

#if USE_TDM
        if (wv == 0 && c + 1 < nch) __builtin_amdgcn_s_wait_tensorcnt(0);
#endif
        __syncthreads();
        cur ^= 1;
    }

    // ---- epilogue: bias + optional residual + optional leaky-relu (both subtiles)
    const int mhi = (lane >> 4) * 8;     // D: lanes 16-31 hold rows M=8..15
    const int q0  = Q0 + xs + nB;
    #pragma unroll
    for (int r8 = 0; r8 < 8; ++r8) {
        int co = co0 + mhi + r8;
        float b = bias[co];
        float v0 = acc0[r8] + b;
        float v1 = acc1[r8] + b;
        if (res) {
            const float* rp = res + (long)bz * res_bstride + (long)co * HW;
            v0 += rp[q0];
            v1 += rp[q0 + 16];
        }
        if (leaky) {
            v0 = (v0 >= 0.0f) ? v0 : 0.1f * v0;
            v1 = (v1 >= 0.0f) ? v1 : 0.1f * v1;
        }
        long oi = (long)bz * out_bstride + (long)co * HW + q0;
        if (outf)  { outf[oi]  = v0;          outf[oi + 16]  = v1; }
        if (outbf) { outbf[oi] = (__bf16)v0;  outbf[oi + 16] = (__bf16)v1; }
    }
}

// ---------------------------------------------------------------------------
__global__ void k_cvt_bf16(const float* __restrict__ in, __bf16* __restrict__ out, long n)
{
    long i = (long)blockIdx.x * blockDim.x + threadIdx.x;
    if (i < n) out[i] = (__bf16)in[i];
}

// ---------------------------------------------------------------------------
// q_in = concat(warp01, frame[:,1], warp21, flow_fwd[:,1], flow_bwd[:,0]) bf16
// ---------------------------------------------------------------------------
__global__ void k_warp_concat(const float* __restrict__ frame,
                              const float* __restrict__ ffw,
                              const float* __restrict__ fbw,
                              __bf16* __restrict__ qin,
                              int H, int W, long total)
{
    long i = (long)blockIdx.x * blockDim.x + threadIdx.x;
    if (i >= total) return;
    const int HW = H * W;
    int q  = (int)(i % HW);
    long t = i / HW;
    int ch = (int)(t % 100);
    int bz = (int)(t / 100);
    int x = q % W, y = q / W;
    float val;
    if (ch >= 96) {
        int j = ch - 96;  // 0,1: flow_fwd[:,1] ; 2,3: flow_bwd[:,0]
        val = (j < 2) ? ffw[((long)(bz * 2 + 1) * 2 + j) * HW + q]
                      : fbw[((long)(bz * 2 + 0) * 2 + (j - 2)) * HW + q];
    } else if (ch >= 32 && ch < 64) {
        val = frame[((long)(bz * 3 + 1) * 32 + (ch - 32)) * HW + q];
    } else {
        int srcT = (ch < 32) ? 0 : 2;
        int c    = (ch < 32) ? ch : ch - 64;
        float fx, fy;
        if (srcT == 0) {
            fx = fbw[((long)(bz * 2 + 0) * 2 + 0) * HW + q];
            fy = fbw[((long)(bz * 2 + 0) * 2 + 1) * HW + q];
        } else {
            fx = ffw[((long)(bz * 2 + 1) * 2 + 0) * HW + q];
            fy = ffw[((long)(bz * 2 + 1) * 2 + 1) * HW + q];
        }
        const float* img = frame + ((long)(bz * 3 + srcT) * 32 + c) * HW;
        float gx = x + fx, gy = y + fy;
        float x0f = floorf(gx), y0f = floorf(gy);
        float wx = gx - x0f, wy = gy - y0f;
        int ix = (int)x0f, iy = (int)y0f;
        auto g = [&](int xi, int yi) -> float {
            xi = min(max(xi, 0), W - 1);
            yi = min(max(yi, 0), H - 1);
            return img[(long)yi * W + xi];
        };
        val = g(ix, iy)         * (1.f - wx) * (1.f - wy)
            + g(ix + 1, iy)     * wx         * (1.f - wy)
            + g(ix, iy + 1)     * (1.f - wx) * wy
            + g(ix + 1, iy + 1) * wx         * wy;
    }
    qin[i] = (__bf16)val;
}

// ---------------------------------------------------------------------------
__global__ void k_softmax12(const float* __restrict__ awr, float* __restrict__ attn,
                            int HW, long total)
{
    long i = (long)blockIdx.x * blockDim.x + threadIdx.x;
    if (i >= total) return;  // total = B*4*HW (q fastest)
    int q  = (int)(i % HW);
    long t = i / HW;
    int m  = (int)(t % 4);
    int bz = (int)(t / 4);
    const float* src = awr + ((long)bz * 48 + m * 12) * HW + q;
    float v[12], mx = -1e30f;
    #pragma unroll
    for (int j = 0; j < 12; ++j) { v[j] = src[(long)j * HW]; mx = fmaxf(mx, v[j]); }
    float s = 0.0f;
    #pragma unroll
    for (int j = 0; j < 12; ++j) { v[j] = __expf(v[j] - mx); s += v[j]; }
    float inv = 1.0f / s;
    float* dst = attn + ((long)bz * 48 + m * 12) * HW + q;
    #pragma unroll
    for (int j = 0; j < 12; ++j) dst[(long)j * HW] = v[j] * inv;
}

// ---------------------------------------------------------------------------
// Deformable sampling + attention-weighted reduction (zero-pad bilinear).
// One thread per (b, m, q), accumulating D=8 channels over L=3, P=4.
// ---------------------------------------------------------------------------
__global__ void k_deform(const float* __restrict__ vimg, const float* __restrict__ off,
                         const float* __restrict__ attn, const float* __restrict__ ffw,
                         const float* __restrict__ fbw, __bf16* __restrict__ outbf,
                         int H, int W, long total)
{
    long i = (long)blockIdx.x * blockDim.x + threadIdx.x;
    if (i >= total) return;  // total = B*4*HW (q fastest)
    const int HW = H * W;
    int q  = (int)(i % HW);
    long t = i / HW;
    int m  = (int)(t % 4);
    int bz = (int)(t / 4);
    int x = q % W, y = q / W;

    float acc[8] = {0.f, 0.f, 0.f, 0.f, 0.f, 0.f, 0.f, 0.f};

    #pragma unroll
    for (int l = 0; l < 3; ++l) {
        float fx = 0.f, fy = 0.f;
        if (l == 0) {
            fx = fbw[((long)(bz * 2 + 0) * 2 + 0) * HW + q];
            fy = fbw[((long)(bz * 2 + 0) * 2 + 1) * HW + q];
        } else if (l == 2) {
            fx = ffw[((long)(bz * 2 + 1) * 2 + 0) * HW + q];
            fy = ffw[((long)(bz * 2 + 1) * 2 + 1) * HW + q];
        }
        const float* img = vimg + ((long)(bz * 3 + l) * 32 + m * 8) * HW;
        #pragma unroll
        for (int p = 0; p < 4; ++p) {
            int ch = ((m * 3 + l) * 4 + p) * 2;
            float ox = off[((long)bz * 96 + ch)     * HW + q] + fx;
            float oy = off[((long)bz * 96 + ch + 1) * HW + q] + fy;
            float aw = attn[((long)bz * 48 + m * 12 + l * 4 + p) * HW + q];
            float gx = x + ox, gy = y + oy;
            float x0f = floorf(gx), y0f = floorf(gy);
            float wx = gx - x0f, wy = gy - y0f;
            int ix = (int)x0f, iy = (int)y0f;
            float w00 = (1.f - wx) * (1.f - wy) * aw;
            float w10 = wx * (1.f - wy) * aw;
            float w01 = (1.f - wx) * wy * aw;
            float w11 = wx * wy * aw;
            auto tap = [&](int xi, int yi, float wt) {
                if ((unsigned)xi < (unsigned)W && (unsigned)yi < (unsigned)H) {
                    const float* pp = img + (long)yi * W + xi;
                    #pragma unroll
                    for (int d = 0; d < 8; ++d) acc[d] += wt * pp[(long)d * HW];
                }
            };
            tap(ix, iy, w00); tap(ix + 1, iy, w10);
            tap(ix, iy + 1, w01); tap(ix + 1, iy + 1, w11);
        }
    }
    #pragma unroll
    for (int d = 0; d < 8; ++d)
        outbf[((long)bz * 32 + m * 8 + d) * HW + q] = (__bf16)acc[d];
}

// ---------------------------------------------------------------------------
__global__ void k_concat_ffn(const float* __restrict__ out1, const float* __restrict__ srcframe,
                             __bf16* __restrict__ cc, int HW, long total)
{
    long i = (long)blockIdx.x * blockDim.x + threadIdx.x;
    if (i >= total) return;  // total = B*64*HW
    int q  = (int)(i % HW);
    long t = i / HW;
    int ch = (int)(t % 64);
    int bz = (int)(t / 64);
    float v = (ch < 32) ? out1[((long)bz * 32 + ch) * HW + q]
                        : srcframe[((long)(bz * 3 + 1) * 32 + (ch - 32)) * HW + q];
    cc[i] = (__bf16)v;
}

// ---------------------------------------------------------------------------
// Host orchestration
// ---------------------------------------------------------------------------
extern "C" void kernel_launch(void* const* d_in, const int* in_sizes, int n_in,
                              void* d_out, int out_size, void* d_ws, size_t ws_size,
                              hipStream_t stream)
{
    (void)n_in; (void)out_size; (void)ws_size;
    const int H = 256, W = 256, HW = H * W;
    const float* frame    = (const float*)d_in[0];
    const float* srcframe = (const float*)d_in[1];
    const float* ffw      = (const float*)d_in[2];
    const float* fbw      = (const float*)d_in[3];
    const float* qk_w  = (const float*)d_in[4];   const float* qk_b  = (const float*)d_in[5];
    const float* ev_w  = (const float*)d_in[6];   const float* ev_b  = (const float*)d_in[7];
    const float* so_w  = (const float*)d_in[8];   const float* so_b  = (const float*)d_in[9];
    const float* aw_w  = (const float*)d_in[10];  const float* aw_b  = (const float*)d_in[11];
    const float* vp_w  = (const float*)d_in[12];  const float* vp_b  = (const float*)d_in[13];
    const float* op_w  = (const float*)d_in[14];  const float* op_b  = (const float*)d_in[15];
    const float* ff_w  = (const float*)d_in[16];  const float* ff_b  = (const float*)d_in[17];
    const float* f1_w  = (const float*)d_in[18];  const float* f1_b  = (const float*)d_in[19];
    const float* f2_w  = (const float*)d_in[20];  const float* f2_b  = (const float*)d_in[21];
    const float* fu_w  = (const float*)d_in[22];  const float* fu_b  = (const float*)d_in[23];

    const int B = in_sizes[0] / (3 * 32 * HW);   // = 2

    // ---- workspace layout
    char* ws = (char*)d_ws;
    size_t cur = 0;
    auto alloc = [&](size_t bytes) -> void* {
        void* p = ws + cur;
        cur += (bytes + 255) & ~(size_t)255;
        return p;
    };
    __bf16* frame_bf = (__bf16*)alloc((size_t)B * 96  * HW * 2);
    __bf16* qin_bf   = (__bf16*)alloc((size_t)B * 100 * HW * 2);
    __bf16* query_bf = (__bf16*)alloc((size_t)B * 96  * HW * 2);
    __bf16* value_bf = (__bf16*)alloc((size_t)B * 96  * HW * 2);
    float*  v_f32    = (float*) alloc((size_t)B * 96  * HW * 4);
    float*  off_f32  = (float*) alloc((size_t)B * 96  * HW * 4);
    float*  awr_f32  = (float*) alloc((size_t)B * 48  * HW * 4);
    float*  attn_f32 = (float*) alloc((size_t)B * 48  * HW * 4);
    __bf16* dout_bf  = (__bf16*)alloc((size_t)B * 32  * HW * 2);
    __bf16* t1_bf    = (__bf16*)alloc((size_t)B * 32  * HW * 2);
    float*  out1_f32 = (float*) alloc((size_t)B * 32  * HW * 4);
    __bf16* cc_bf    = (__bf16*)alloc((size_t)B * 64  * HW * 2);
    __bf16* ffn_bf   = (__bf16*)alloc((size_t)B * 32  * HW * 2);
    __bf16* out2_bf  = (__bf16*)alloc((size_t)B * 32  * HW * 2);

    auto gsz = [](long n) { return dim3((unsigned)((n + 255) / 256)); };

    // Packed bf16 weights (tile layout); pack() allocates + converts.
    auto pack = [&](const float* w, int Cin, int Cout, int KH, int KW) -> __bf16* {
        int nch = (Cin + 31) >> 5, taps = KH * KW;
        long n = (long)Cout * nch * taps * 32;
        __bf16* p = (__bf16*)alloc((size_t)n * 2);
        k_pack_weights<<<gsz(n), 256, 0, stream>>>(w, p, Cin, Cout, KH, KW, n);
        return p;
    };
    __bf16* qk_p = pack(qk_w, 100, 96, 3, 3);
    __bf16* ev_p = pack(ev_w,  96, 96, 3, 3);
    __bf16* so_p = pack(so_w,  96, 96, 3, 3);
    __bf16* aw_p = pack(aw_w,  96, 48, 3, 3);
    __bf16* vp_p = pack(vp_w,  32, 32, 3, 3);
    __bf16* op_p = pack(op_w,  32, 32, 3, 3);
    __bf16* ff_p = pack(ff_w,  32, 32, 3, 3);
    __bf16* f1_p = pack(f1_w,  64, 32, 3, 3);
    __bf16* f2_p = pack(f2_w,  32, 32, 3, 3);
    __bf16* fu_p = pack(fu_w,  32, 32, 1, 1);

    auto conv3 = [&](const __bf16* in, int Cin, long in_bs,
                     const __bf16* wp, const float* b, int Cout, int pad, int dil,
                     float* outf, __bf16* outbf, long out_bs,
                     const float* res, long res_bs, int leaky, int batches) {
        dim3 g(HW / 256, Cout / 16, batches);
        conv_wmma_kernel<3, 3><<<g, dim3(256), 0, stream>>>(
            in, Cin, in_bs, wp, b, Cout, (Cin + 31) >> 5, pad, dil,
            outf, outbf, out_bs, res, res_bs, H, W, leaky);
    };
    auto conv1 = [&](const __bf16* in, int Cin, long in_bs,
                     const __bf16* wp, const float* b, int Cout,
                     float* outf, __bf16* outbf, long out_bs,
                     const float* res, long res_bs, int leaky, int batches) {
        dim3 g(HW / 256, Cout / 16, batches);
        conv_wmma_kernel<1, 1><<<g, dim3(256), 0, stream>>>(
            in, Cin, in_bs, wp, b, Cout, (Cin + 31) >> 5, 0, 1,
            outf, outbf, out_bs, res, res_bs, H, W, leaky);
    };

    // 1) frame -> bf16 (emb_v input, (b,96,HW))
    long nFrame = (long)B * 96 * HW;
    k_cvt_bf16<<<gsz(nFrame), 256, 0, stream>>>(frame, frame_bf, nFrame);

    // 2) warp + concat -> q_in bf16 (b,100,HW)
    long nQin = (long)B * 100 * HW;
    k_warp_concat<<<gsz(nQin), 256, 0, stream>>>(frame, ffw, fbw, qin_bf, H, W, nQin);

    // 3) query = lrelu(conv(q_in, emb_qk))
    conv3(qin_bf, 100, (long)100 * HW, qk_p, qk_b, 96, 1, 1,
          nullptr, query_bf, (long)96 * HW, nullptr, 0, 1, B);

    // 4) value = lrelu(conv(frame, emb_v))
    conv3(frame_bf, 96, (long)96 * HW, ev_p, ev_b, 96, 1, 1,
          nullptr, value_bf, (long)96 * HW, nullptr, 0, 1, B);

    // 5) v = conv(value, vp): 6 batches of (32,HW)
    conv3(value_bf, 32, (long)32 * HW, vp_p, vp_b, 32, 1, 1,
          v_f32, nullptr, (long)32 * HW, nullptr, 0, 0, B * 3);

    // 6) off = conv(query, so)
    conv3(query_bf, 96, (long)96 * HW, so_p, so_b, 96, 1, 1,
          off_f32, nullptr, (long)96 * HW, nullptr, 0, 0, B);

    // 7) aw raw = conv(query, aw)
    conv3(query_bf, 96, (long)96 * HW, aw_p, aw_b, 48, 1, 1,
          awr_f32, nullptr, (long)48 * HW, nullptr, 0, 0, B);

    // 8) softmax over 12
    long nSm = (long)B * 4 * HW;
    k_softmax12<<<gsz(nSm), 256, 0, stream>>>(awr_f32, attn_f32, HW, nSm);

    // 9) deformable sampling + einsum -> (b,32,HW) bf16
    k_deform<<<gsz(nSm), 256, 0, stream>>>(v_f32, off_f32, attn_f32, ffw, fbw,
                                           dout_bf, H, W, nSm);

    // 10) op conv
    conv3(dout_bf, 32, (long)32 * HW, op_p, op_b, 32, 1, 1,
          nullptr, t1_bf, (long)32 * HW, nullptr, 0, 0, B);

    // 11) ff conv + residual frame[:,1]
    conv3(t1_bf, 32, (long)32 * HW, ff_p, ff_b, 32, 1, 1,
          out1_f32, nullptr, (long)32 * HW, frame + (long)32 * HW, (long)96 * HW, 0, B);

    // 12) concat(out1, srcframe[:,1]) -> bf16 (b,64,HW)
    long nCc = (long)B * 64 * HW;
    k_concat_ffn<<<gsz(nCc), 256, 0, stream>>>(out1_f32, srcframe, cc_bf, HW, nCc);

    // 13) ffn1: dilated conv (pad=2, dil=2) + lrelu
    conv3(cc_bf, 64, (long)64 * HW, f1_p, f1_b, 32, 2, 2,
          nullptr, ffn_bf, (long)32 * HW, nullptr, 0, 1, B);

    // 14) ffn2 conv + residual out1
    conv3(ffn_bf, 32, (long)32 * HW, f2_p, f2_b, 32, 1, 1,
          nullptr, out2_bf, (long)32 * HW, out1_f32, (long)32 * HW, 0, B);

    // 15) fus 1x1 conv + lrelu -> final f32 output
    conv1(out2_bf, 32, (long)32 * HW, fu_p, fu_b, 32,
          (float*)d_out, nullptr, (long)32 * HW, nullptr, 0, 1, B);
}
